// MultiHeadAttention_68058051772537
// MI455X (gfx1250) — matrix operand order repaired and checked by reading, not compiled, not run on hardware
//
#include <hip/hip_runtime.h>
#include <hip/hip_bf16.h>

// ---------------------------------------------------------------------------
// MHA for MI455X (gfx1250, wave32, WMMA f32_16x16x32_f16).
// Roofline: the 1 GiB attn output dominates. Flash-style two-pass attention
// in transposed orientation (keys=M, queries=N) makes softmax stats lane-
// local and lets ctx = P@V fuse into the same loop: attn matrix is written
// exactly once (non-temporal) and never re-read. Total attn traffic: 1 GiB
// (vs 4 GiB for the naive score->softmax->ctx pipeline); extra score
// recompute is only ~34 GFLOP of f16 WMMA.
// ---------------------------------------------------------------------------

typedef __attribute__((ext_vector_type(16))) _Float16 v16h;
typedef __attribute__((ext_vector_type(8)))  _Float16 v8h;
typedef __attribute__((ext_vector_type(8)))  float    v8f;
typedef __attribute__((ext_vector_type(4)))  float    v4f;
typedef __attribute__((ext_vector_type(4)))  int      v4i;

#define DMODEL 1024
#define SEQ    2048
#define BATCH  4
#define NHEAD  16
#define DKH    64
#define NTOK   (BATCH * SEQ)    // 8192 tokens
#define NBH    (BATCH * NHEAD)  // 64 (b,h) pairs

// ---------------- fragment helpers (contiguous K: lane half h holds k 16h..16h+15) ----------

__device__ __forceinline__ v16h cat8(v8h lo, v8h hi) {
  return __builtin_shufflevector(lo, hi, 0, 1, 2, 3, 4, 5, 6, 7,
                                         8, 9, 10, 11, 12, 13, 14, 15);
}

__device__ __forceinline__ v8h ld8h(const _Float16* p) {
  return *(const v8h*)p;  // global_load_b128
}
__device__ __forceinline__ v8h ld8h(const float* p) {  // f32 source, cvt to f16
  v4f a = *(const v4f*)p;
  v4f b = *(const v4f*)(p + 4);
  v8h r;
#pragma unroll
  for (int i = 0; i < 4; ++i) { r[i] = (_Float16)a[i]; r[i + 4] = (_Float16)b[i]; }
  return r;
}

// A fragment, 16x32 (MxK) 16-bit. Lane l holds row M=l&15; pointer must be
// pre-offset by row*lda + (l>=16?16:0); 16 contiguous K.
template <typename T>
__device__ __forceinline__ v16h load_fragA(const T* p) {
  return cat8(ld8h(p), ld8h(p + 8));
}
// B fragment, 32x16 (KxN), B stored row-major [N x K] (NT gemm). Lane l holds
// column N=l&15; pointer pre-offset by col*ldb + (l>=16?16:0); 16 contiguous K.
template <typename T>
__device__ __forceinline__ v16h load_fragB(const T* p) {
  return cat8(ld8h(p), ld8h(p + 8));
}

__device__ __forceinline__ v8f wmma16(v16h a, v16h b, v8f c) {
  return __builtin_amdgcn_wmma_f32_16x16x32_f16(false, a, false, b,
                                                (short)0, c, false, false);
}

// ---------------- wave-level 32x32 NT-GEMM (proj / outproj) ----------------
template <typename TA, typename TB>
__device__ __forceinline__ void gemm_nt_32x32(const TA* __restrict__ A, size_t lda,
                                              const TB* __restrict__ B, size_t ldb,
                                              int K, v8f acc[2][2]) {
  const int lane = threadIdx.x & 31;
  const int r16  = lane & 15;
  const int ao   = (lane >> 4) * 16;
  const int bo   = (lane >> 4) * 16;
  const TA* a0 = A + (size_t)r16 * lda + ao;
  const TA* a1 = a0 + (size_t)16 * lda;
  const TB* b0 = B + (size_t)r16 * ldb + bo;
  const TB* b1 = b0 + (size_t)16 * ldb;

  for (int k = 0; k < K; k += 32) {
    if (k + 128 < K) {  // global_prefetch_b8
      __builtin_prefetch((const void*)(a0 + k + 128), 0, 0);
      __builtin_prefetch((const void*)(b0 + k + 128), 0, 0);
    }
    v16h fa0 = load_fragA(a0 + k);
    v16h fa1 = load_fragA(a1 + k);
    v16h fb0 = load_fragB(b0 + k);
    v16h fb1 = load_fragB(b1 + k);
    acc[0][0] = wmma16(fa0, fb0, acc[0][0]);
    acc[0][1] = wmma16(fa0, fb1, acc[0][1]);
    acc[1][0] = wmma16(fa1, fb0, acc[1][0]);
    acc[1][1] = wmma16(fa1, fb1, acc[1][1]);
  }
}

// Workgroup = 256 threads = 8 waves. proj/outproj: 4(M) x 2(N) -> 128x64 tile.

// ---------------- kernel 1: QKV projections -------------------------------
__global__ __launch_bounds__(256) void proj_kernel(
    const float* __restrict__ q, const float* __restrict__ k,
    const float* __restrict__ v,
    const float* __restrict__ Wq, const float* __restrict__ bq,
    const float* __restrict__ Wk, const float* __restrict__ bk,
    const float* __restrict__ Wv, const float* __restrict__ bv,
    _Float16* __restrict__ Qh, _Float16* __restrict__ Kh,
    _Float16* __restrict__ Vt) {
  const int z = blockIdx.z;
  const float* X    = (z == 0) ? q  : (z == 1) ? k  : v;
  const float* W    = (z == 0) ? Wq : (z == 1) ? Wk : Wv;
  const float* bias = (z == 0) ? bq : (z == 1) ? bk : bv;

  const int wave = threadIdx.x >> 5, lane = threadIdx.x & 31;
  const int row0 = blockIdx.y * 128 + (wave >> 1) * 32;
  const int col0 = blockIdx.x * 64 + (wave & 1) * 32;

  v8f acc[2][2] = {};
  gemm_nt_32x32(X + (size_t)row0 * DMODEL, DMODEL,
                W + (size_t)col0 * DMODEL, DMODEL, DMODEL, acc);

  const int cn = lane & 15, rb = (lane >> 4) * 8;
#pragma unroll
  for (int i = 0; i < 2; ++i)
#pragma unroll
    for (int j = 0; j < 2; ++j) {
      const int col = col0 + j * 16 + cn;
      const float bi = bias[col];
      const int h = col >> 6, dk = col & 63;
#pragma unroll
      for (int r = 0; r < 8; ++r) {
        const int row = row0 + i * 16 + rb + r;
        const int b = row >> 11, s = row & (SEQ - 1);
        const size_t bh = (size_t)(b * NHEAD + h);
        const float val = acc[i][j][r] + bi;
        if (z == 0)
          Qh[(bh * SEQ + s) * DKH + dk] = (_Float16)(val * 0.125f);  // 1/sqrt(dk)
        else if (z == 1)
          Kh[(bh * SEQ + s) * DKH + dk] = (_Float16)val;
        else
          Vt[(bh * DKH + dk) * SEQ + s] = (_Float16)val;  // transposed
      }
    }
}

// ------- attention tile mapping (both passes) ------------------------------
// ST[key][q] tiles: keys = M (rows, in accumulator VGPRs), q = N (lanes).
// WG = 8 waves = 4 wave-pairs; pair wp owns 32 queries; wn in {0,1} splits
// the 2048 keys into interleaved 32-key chunks. Q B-fragments are loop-
// invariant (loaded once).

// ---------------- kernel 2: softmax statistics (m, l) per query row -------
__global__ __launch_bounds__(256) void attn_stats_kernel(
    const _Float16* __restrict__ Qh, const _Float16* __restrict__ Kh,
    float2* __restrict__ stats) {
  __shared__ float sm[2][128], sl[2][128];
  const int bh = blockIdx.y;
  const int wave = threadIdx.x >> 5, lane = threadIdx.x & 31;
  const int wn = wave & 1, wp = wave >> 1;
  const int qb = blockIdx.x * 128 + wp * 32;
  const int l16 = lane & 15;
  const int ao = (lane >> 4) * 16;
  const int bo = (lane >> 4) * 16;
  const _Float16* Qb = Qh + (size_t)bh * SEQ * DKH;
  const _Float16* Kb = Kh + (size_t)bh * SEQ * DKH;

  v16h fq[2][2];  // [qfrag j][kstep] - loop invariant
#pragma unroll
  for (int j = 0; j < 2; ++j)
#pragma unroll
    for (int ks = 0; ks < 2; ++ks)
      fq[j][ks] = load_fragB(Qb + (size_t)(qb + j * 16 + l16) * DKH + ks * 32 + bo);

  float m[2] = {-3.0e38f, -3.0e38f};
  float l[2] = {0.f, 0.f};

  for (int it = 0; it < SEQ / 64; ++it) {
    const int kb = it * 64 + wn * 32;
    v8f acc[2][2] = {};
#pragma unroll
    for (int ks = 0; ks < 2; ++ks) {
      v16h fk0 = load_fragA(Kb + (size_t)(kb + l16) * DKH + ks * 32 + ao);
      v16h fk1 = load_fragA(Kb + (size_t)(kb + 16 + l16) * DKH + ks * 32 + ao);
      acc[0][0] = wmma16(fk0, fq[0][ks], acc[0][0]);
      acc[0][1] = wmma16(fk0, fq[1][ks], acc[0][1]);
      acc[1][0] = wmma16(fk1, fq[0][ks], acc[1][0]);
      acc[1][1] = wmma16(fk1, fq[1][ks], acc[1][1]);
    }
    // lane-local online softmax update over this lane's 16 keys per q col
#pragma unroll
    for (int j = 0; j < 2; ++j) {
      float t = acc[0][j][0];
#pragma unroll
      for (int r = 1; r < 8; ++r) t = fmaxf(t, acc[0][j][r]);
#pragma unroll
      for (int r = 0; r < 8; ++r) t = fmaxf(t, acc[1][j][r]);
      const float nm = fmaxf(m[j], t);
      float s = 0.f;
#pragma unroll
      for (int r = 0; r < 8; ++r) s += __expf(acc[0][j][r] - nm);
#pragma unroll
      for (int r = 0; r < 8; ++r) s += __expf(acc[1][j][r] - nm);
      l[j] = l[j] * __expf(m[j] - nm) + s;
      m[j] = nm;
    }
  }
  // merge the two lane-halves (they hold different key rows of the same q)
#pragma unroll
  for (int j = 0; j < 2; ++j) {
    const float pm = __shfl_xor(m[j], 16, 32);
    const float pl = __shfl_xor(l[j], 16, 32);
    const float nm = fmaxf(m[j], pm);
    l[j] = l[j] * __expf(m[j] - nm) + pl * __expf(pm - nm);
    m[j] = nm;
  }
  if (lane < 16) {
#pragma unroll
    for (int j = 0; j < 2; ++j) {
      sm[wn][wp * 32 + j * 16 + lane] = m[j];
      sl[wn][wp * 32 + j * 16 + lane] = l[j];
    }
  }
  __syncthreads();
  const int t = threadIdx.x;
  if (t < 128) {  // merge the two key-subset waves of each pair
    const float m0 = sm[0][t], m1 = sm[1][t];
    const float nm = fmaxf(m0, m1);
    const float L = sl[0][t] * __expf(m0 - nm) + sl[1][t] * __expf(m1 - nm);
    stats[(size_t)bh * SEQ + blockIdx.x * 128 + t] = make_float2(nm, L);
  }
}

// ---------------- kernel 3: fused attn write + ctx = P @ V ----------------
__global__ __launch_bounds__(256) void attn_ctx_kernel(
    const _Float16* __restrict__ Qh, const _Float16* __restrict__ Kh,
    const _Float16* __restrict__ Vt, const float2* __restrict__ stats,
    float* __restrict__ attn, _Float16* __restrict__ ctx) {
  __shared__ float cmerge[4][32][64];  // [wave-pair][q_local][dk] = 32 KB
  const int bh = blockIdx.y;
  const int b = bh >> 4, h = bh & 15;
  const int wave = threadIdx.x >> 5, lane = threadIdx.x & 31;
  const int wn = wave & 1, wp = wave >> 1;
  const int qb = blockIdx.x * 128 + wp * 32;
  const int l16 = lane & 15;
  const int ao = (lane >> 4) * 16;
  const int bo = (lane >> 4) * 16;
  const int rb = (lane >> 4) * 8;
  const _Float16* Qb = Qh + (size_t)bh * SEQ * DKH;
  const _Float16* Kb = Kh + (size_t)bh * SEQ * DKH;
  const _Float16* Vb = Vt + (size_t)bh * DKH * SEQ;
  float* attnq = attn + (size_t)bh * SEQ * SEQ;

  v16h fq[2][2];
  float mj[2], invl[2];
#pragma unroll
  for (int j = 0; j < 2; ++j) {
#pragma unroll
    for (int ks = 0; ks < 2; ++ks)
      fq[j][ks] = load_fragB(Qb + (size_t)(qb + j * 16 + l16) * DKH + ks * 32 + bo);
    const float2 st = stats[(size_t)bh * SEQ + qb + j * 16 + l16];
    mj[j] = st.x;
    invl[j] = 1.0f / st.y;
  }

  v8f cacc[4][2] = {};  // ctx^T frags: [dk group][q frag]

  for (int it = 0; it < SEQ / 64; ++it) {
    const int kb = it * 64 + wn * 32;
    // --- recompute 32x32 score tile ---
    v8f acc[2][2] = {};
#pragma unroll
    for (int ks = 0; ks < 2; ++ks) {
      v16h fk0 = load_fragA(Kb + (size_t)(kb + l16) * DKH + ks * 32 + ao);
      v16h fk1 = load_fragA(Kb + (size_t)(kb + 16 + l16) * DKH + ks * 32 + ao);
      acc[0][0] = wmma16(fk0, fq[0][ks], acc[0][0]);
      acc[0][1] = wmma16(fk0, fq[1][ks], acc[0][1]);
      acc[1][0] = wmma16(fk1, fq[0][ks], acc[1][0]);
      acc[1][1] = wmma16(fk1, fq[1][ks], acc[1][1]);
    }
    // --- normalize, write attn once (keys contiguous per lane), pack f16 ---
    v8h pk[2][2];  // [q frag j][key frag i]
#pragma unroll
    for (int j = 0; j < 2; ++j)
#pragma unroll
      for (int i = 0; i < 2; ++i) {
        v4f p0, p1;
#pragma unroll
        for (int r = 0; r < 4; ++r)
          p0[r] = __expf(acc[i][j][r] - mj[j]) * invl[j];
#pragma unroll
        for (int r = 0; r < 4; ++r)
          p1[r] = __expf(acc[i][j][4 + r] - mj[j]) * invl[j];
        float* dst = attnq + (size_t)(qb + j * 16 + l16) * SEQ + kb + i * 16 + rb;
        __builtin_nontemporal_store(p0, (v4f*)dst);
        __builtin_nontemporal_store(p1, (v4f*)(dst + 4));
        v8h ph;
#pragma unroll
        for (int r = 0; r < 4; ++r) {
          ph[r] = (_Float16)p0[r];
          ph[4 + r] = (_Float16)p1[r];
        }
        pk[j][i] = ph;
      }
    // --- build P^T B-fragments (one shfl_xor(16) exchange) and ctx WMMA ---
#pragma unroll
    for (int j = 0; j < 2; ++j) {
      v8h send = (lane < 16) ? pk[j][1] : pk[j][0];
      v4i si = __builtin_bit_cast(v4i, send);
#pragma unroll
      for (int w = 0; w < 4; ++w) si[w] = __shfl_xor(si[w], 16, 32);
      const v8h recv = __builtin_bit_cast(v8h, si);
      const v16h bp = (lane < 16) ? cat8(pk[j][0], recv) : cat8(recv, pk[j][1]);
#pragma unroll
      for (int di = 0; di < 4; ++di) {
        v16h fv = load_fragA(Vb + (size_t)(di * 16 + l16) * SEQ + kb + ao);
        cacc[di][j] = wmma16(fv, bp, cacc[di][j]);
      }
    }
  }

  // --- merge the two key-subset waves of each pair, store ctx f16 ---------
  if (wn == 1) {
#pragma unroll
    for (int di = 0; di < 4; ++di)
#pragma unroll
      for (int j = 0; j < 2; ++j) {
        float* d = &cmerge[wp][j * 16 + l16][di * 16 + rb];
        v4f lo, hi;
#pragma unroll
        for (int r = 0; r < 4; ++r) { lo[r] = cacc[di][j][r]; hi[r] = cacc[di][j][4 + r]; }
        *(v4f*)d = lo;
        *(v4f*)(d + 4) = hi;
      }
  }
  __syncthreads();
  if (wn == 0) {
#pragma unroll
    for (int di = 0; di < 4; ++di)
#pragma unroll
      for (int j = 0; j < 2; ++j) {
        const float* d = &cmerge[wp][j * 16 + l16][di * 16 + rb];
        v8h oh;
#pragma unroll
        for (int r = 0; r < 8; ++r) oh[r] = (_Float16)(cacc[di][j][r] + d[r]);
        const int qg = qb + j * 16 + l16;
        *(v8h*)(ctx + ((size_t)b * SEQ + qg) * DMODEL + h * DKH + di * 16 + rb) = oh;
      }
  }
}

// ---------------- kernel 4: out = ctx @ Wo^T + bo -------------------------
__global__ __launch_bounds__(256) void outproj_kernel(
    const _Float16* __restrict__ ctx, const float* __restrict__ Wo,
    const float* __restrict__ bo, float* __restrict__ out) {
  const int wave = threadIdx.x >> 5, lane = threadIdx.x & 31;
  const int row0 = blockIdx.y * 128 + (wave >> 1) * 32;
  const int col0 = blockIdx.x * 64 + (wave & 1) * 32;

  v8f acc[2][2] = {};
  gemm_nt_32x32(ctx + (size_t)row0 * DMODEL, DMODEL,
                Wo + (size_t)col0 * DMODEL, DMODEL, DMODEL, acc);

  const int cn = lane & 15, rb = (lane >> 4) * 8;
#pragma unroll
  for (int i = 0; i < 2; ++i)
#pragma unroll
    for (int j = 0; j < 2; ++j) {
      const int col = col0 + j * 16 + cn;
      const float bi = bo[col];
#pragma unroll
      for (int r = 0; r < 8; ++r) {
        const int row = row0 + i * 16 + rb + r;
        out[(size_t)row * DMODEL + col] = acc[i][j][r] + bi;
      }
    }
}

// ---------------------------------------------------------------------------

extern "C" void kernel_launch(void* const* d_in, const int* in_sizes, int n_in,
                              void* d_out, int out_size, void* d_ws, size_t ws_size,
                              hipStream_t stream) {
  const float* q  = (const float*)d_in[0];
  const float* k  = (const float*)d_in[1];
  const float* v  = (const float*)d_in[2];
  const float* Wq = (const float*)d_in[3];
  const float* bq = (const float*)d_in[4];
  const float* Wk = (const float*)d_in[5];
  const float* bk = (const float*)d_in[6];
  const float* Wv = (const float*)d_in[7];
  const float* bv = (const float*)d_in[8];
  const float* Wo = (const float*)d_in[9];
  const float* bo = (const float*)d_in[10];

  float* out  = (float*)d_out;
  float* attn = out + (size_t)NTOK * DMODEL;  // second tuple output, 1 GiB

  const size_t headElems = (size_t)NBH * SEQ * DKH;  // 8,388,608
  _Float16* Qh  = (_Float16*)d_ws;
  _Float16* Kh  = Qh + headElems;
  _Float16* Vt  = Kh + headElems;
  _Float16* ctx = Vt + headElems;                 // [B,S,D] f16
  float2*   stats = (float2*)(ctx + headElems);   // [NBH*SEQ] (m, l) -> +1 MiB

  const dim3 blk(256);

  // 1) Q/K/V projections (Q pre-scaled by 1/sqrt(dk), V stored transposed)
  proj_kernel<<<dim3(DMODEL / 64, NTOK / 128, 3), blk, 0, stream>>>(
      q, k, v, Wq, bq, Wk, bk, Wv, bv, Qh, Kh, Vt);

  // 2) softmax row statistics (scores recomputed, never stored)
  attn_stats_kernel<<<dim3(SEQ / 128, NBH), blk, 0, stream>>>(Qh, Kh, stats);

  // 3) fused: normalized attn written once (NT) + ctx accumulation
  attn_ctx_kernel<<<dim3(SEQ / 128, NBH), blk, 0, stream>>>(Qh, Kh, Vt, stats,
                                                            attn, ctx);

  // 4) output projection
  outproj_kernel<<<dim3(DMODEL / 64, NTOK / 128, 1), blk, 0, stream>>>(
      ctx, Wo, bo, out);
}
